// HGraphormer_33569464386365
// MI455X (gfx1250) — matrix operand order repaired and checked by reading, not compiled, not run on hardware
//
#include <hip/hip_runtime.h>

// ---------------------------------------------------------------------------
// HGraphormer forward for MI455X (gfx1250, wave32, WMMA bf16 16x16x32 + TDM)
// ---------------------------------------------------------------------------
typedef __bf16 bf16;
typedef __attribute__((ext_vector_type(16))) __bf16 v16bf;
typedef __attribute__((ext_vector_type(8)))  float  v8f;
typedef __attribute__((ext_vector_type(4)))  unsigned int u32x4;
typedef __attribute__((ext_vector_type(8)))  int  i32x8;
typedef __attribute__((ext_vector_type(4)))  int  i32x4;

#define N_NODES 4096
#define E_EDGES 4096
#define NHID    512
#define NHEAD   8
#define DKH     64
#define NLAYER  4
#define NCLASS  16

#if defined(__has_builtin)
#if __has_builtin(__builtin_amdgcn_tensor_load_to_lds) && \
    __has_builtin(__builtin_amdgcn_s_wait_tensorcnt)
#define HAVE_TDM 1
#endif
#endif
#ifndef HAVE_TDM
#define HAVE_TDM 0
#endif

#if HAVE_TDM
// Issue a TDM 2-D tile load: global (row-major, pitch stride_elems bf16)
// -> LDS (tile stored contiguously [tile_y][tile_x]).  Per-wave instruction;
// call from one wave only, then s_wait_tensorcnt + workgroup barrier.
// D# layout per CDNA5 ISA 8.3/8.4 (group0: count/lds/global/type,
// group1: data_size=2B, dims, tile dims, dim0 stride).
__device__ __forceinline__ void tdm_load_2d(const bf16* gsrc, unsigned lds_off,
                                            int tile_x, int tile_y,
                                            int stride_elems) {
    const unsigned long long ga = (unsigned long long)(uintptr_t)gsrc;
    u32x4 g0;
    g0[0] = 1u;                                            // count=1 (valid)
    g0[1] = lds_off;                                       // lds_addr (bytes)
    g0[2] = (unsigned)(ga & 0xFFFFFFFFu);                  // global_addr[31:0]
    g0[3] = (unsigned)((ga >> 32) & 0x1FFFFFFu)            // global_addr[56:32]
          | (2u << 30);                                    // type = 2 ("image")
    i32x8 g1;
    g1[0] = 1 << 16;                                       // data_size = 1 -> 2B
    g1[1] = (tile_x & 0xFFFF) << 16;                       // tensor_dim0[15:0]
    g1[2] = (tile_y & 0xFFFF) << 16;                       // tensor_dim1[15:0]
    g1[3] = (tile_x & 0xFFFF) << 16;                       // tile_dim0
    g1[4] = tile_y & 0xFFFF;                               // tile_dim1
    g1[5] = stride_elems;                                  // tensor_dim0_stride lo
    g1[6] = 0;                                             // stride hi / dim1 stride
    g1[7] = 0;
    i32x4 z4 = {0, 0, 0, 0};
#if __clang_major__ >= 23
    i32x8 z8 = {0, 0, 0, 0, 0, 0, 0, 0};
    __builtin_amdgcn_tensor_load_to_lds(g0, g1, z4, z4, z8, 0);
#else
    __builtin_amdgcn_tensor_load_to_lds(g0, g1, z4, z4, 0);
#endif
}
#endif

// Build a 16-element bf16 fragment from two 16-byte chunks (ds/global b128s).
__device__ __forceinline__ v16bf ldfrag(const bf16* p0, const bf16* p1) {
    union { uint4 u[2]; v16bf v; } t;
    t.u[0] = *reinterpret_cast<const uint4*>(p0);
    t.u[1] = *reinterpret_cast<const uint4*>(p1);
    return t.v;
}

__device__ __forceinline__ v8f wmma_bf16(v16bf a, v16bf b, v8f c) {
    return __builtin_amdgcn_wmma_f32_16x16x32_bf16(false, a, false, b,
                                                   (short)0, c, false, false);
}

// ---------------------------------------------------------------------------
// Generic bf16 GEMM, f32 accumulate.  C[M,N] = A[M,K] * op(B)  (+bias/scale)
//   TRANSB=1 : B is [N,K] row-major (both operands stream along K)
//   TRANSB=0 : B is [K,N] row-major (transposed while staging into LDS)
// Block: 256 threads = 8 waves; tile 128x128; each wave does 32x64.
// A tile (and B tile when TRANSB) staged by the Tensor Data Mover when
// available; TRANSB=0 B tile is transposed manually into LDS.
// ---------------------------------------------------------------------------
template <bool TRANSB>
__global__ void __launch_bounds__(256)
gemm_bf16_kernel(const bf16* __restrict__ A, const bf16* __restrict__ B,
                 float* __restrict__ Cf, bf16* __restrict__ Cb,
                 int M, int N, int K, int lda, int ldb, int ldc,
                 const float* __restrict__ bias,
                 const float* __restrict__ rscale,
                 const float* __restrict__ cscale) {
    __shared__ bf16 Al[128 * 32];
    __shared__ bf16 Bl[128 * 32];
    const int tid = threadIdx.x;
    const int lane = tid & 31, wave = tid >> 5;
    const int halfl = lane >> 4, r16 = lane & 15;
    const int m0 = blockIdx.y * 128, n0 = blockIdx.x * 128;
    const int wm = wave >> 1, wn = wave & 1;

    const v8f zero = {};
    v8f acc[2][4];
#pragma unroll
    for (int i = 0; i < 2; ++i)
#pragma unroll
        for (int j = 0; j < 4; ++j) acc[i][j] = zero;

    for (int kt = 0; kt < K; kt += 32) {
#if HAVE_TDM
        if (wave == 0) {   // DMA tiles into LDS, overlap with barrier wait
            tdm_load_2d(A + (size_t)m0 * lda + kt,
                        (unsigned)(uintptr_t)(void*)Al, 32, 128, lda);
            if (TRANSB)
                tdm_load_2d(B + (size_t)n0 * ldb + kt,
                            (unsigned)(uintptr_t)(void*)Bl, 32, 128, ldb);
            __builtin_amdgcn_s_wait_tensorcnt(0);
        }
#else
        {   // stage A tile [128 x 32]
            const int row = tid >> 1, seg = tid & 1;
            const uint4* src = reinterpret_cast<const uint4*>(
                A + (size_t)(m0 + row) * lda + kt + seg * 16);
            uint4* dst = reinterpret_cast<uint4*>(Al + row * 32 + seg * 16);
            dst[0] = src[0];
            dst[1] = src[1];
            if (kt + 32 < K)   // hint next K-tile -> global_prefetch_b8
                __builtin_prefetch(
                    (const void*)(A + (size_t)(m0 + row) * lda + kt + 32 + seg * 16), 0, 1);
        }
        if (TRANSB) {          // B[N,K]: copy rows as-is
            const int row = tid >> 1, seg = tid & 1;
            const uint4* src = reinterpret_cast<const uint4*>(
                B + (size_t)(n0 + row) * ldb + kt + seg * 16);
            uint4* dst = reinterpret_cast<uint4*>(Bl + row * 32 + seg * 16);
            dst[0] = src[0];
            dst[1] = src[1];
        }
#endif
        if (!TRANSB) {         // B[K,N]: transpose into Bl[n][k]
            const int krow = tid >> 3, ngrp = (tid & 7) * 16;
            union { uint4 u[2]; bf16 e[16]; } t;
            const uint4* src = reinterpret_cast<const uint4*>(
                B + (size_t)(kt + krow) * ldb + n0 + ngrp);
            t.u[0] = src[0];
            t.u[1] = src[1];
#pragma unroll
            for (int i = 0; i < 16; ++i) Bl[(ngrp + i) * 32 + krow] = t.e[i];
        }
        __syncthreads();

        // A fragments: lane half splits K into 8-elt runs per ISA layout
        const bf16* a0p = Al + (wm * 32 + r16) * 32;
        const bf16* a1p = Al + (wm * 32 + 16 + r16) * 32;
        v16bf a0 = ldfrag(a0p + halfl * 8, a0p + 16 + halfl * 8);
        v16bf a1 = ldfrag(a1p + halfl * 8, a1p + 16 + halfl * 8);
#pragma unroll
        for (int j = 0; j < 4; ++j) {
            const bf16* bp = Bl + (wn * 64 + j * 16 + r16) * 32 + halfl * 8;
            v16bf b = ldfrag(bp, bp + 16);
            acc[0][j] = wmma_bf16(a0, b, acc[0][j]);
            acc[1][j] = wmma_bf16(a1, b, acc[1][j]);
        }
        __syncthreads();
    }

    // Epilogue: C layout = lane column, VGPR r -> row (r + 8*half)
#pragma unroll
    for (int i = 0; i < 2; ++i) {
#pragma unroll
        for (int j = 0; j < 4; ++j) {
            const int col = n0 + wn * 64 + j * 16 + r16;
            const float bc = bias ? bias[col] : 0.0f;
            const float cs = cscale ? cscale[col] : 1.0f;
#pragma unroll
            for (int r = 0; r < 8; ++r) {
                const int row = m0 + wm * 32 + i * 16 + halfl * 8 + r;
                float v = acc[i][j][r];
                if (rscale) v *= rscale[row] * cs;
                v += bc;
                if (Cf) Cf[(size_t)row * ldc + col] = v;
                if (Cb) Cb[(size_t)row * ldc + col] = (bf16)v;
            }
        }
    }
}

// ---------------------------------------------------------------------------
// Flash attention per head with Laplacian blend:
//   ctx = 0.5 * softmax(Q Kh^T / 8) @ Vh + 0.5 * GV   (GV = G@V precomputed)
// Block: 256 threads = 8 waves sharing one head's K/V tiles in LDS;
// each wave owns 16 query rows.  Key chunk = 32.  K tile via TDM when present.
// ---------------------------------------------------------------------------
__global__ void __launch_bounds__(256)
flash_blend_kernel(const bf16* __restrict__ Q, const bf16* __restrict__ Kx,
                   const bf16* __restrict__ V, const float* __restrict__ GV,
                   bf16* __restrict__ ctxb, int n) {
    __shared__ bf16 Kl[32 * 64];        // [key][d]
    __shared__ bf16 Vl[64 * 32];        // [d][key] (transposed)
    __shared__ bf16 Pl[8 * 16 * 32];    // per-wave P tile
    const int tid = threadIdx.x, lane = tid & 31, wave = tid >> 5;
    const int halfl = lane >> 4, r16 = lane & 15;
    const int head = blockIdx.y;
    const int q0 = blockIdx.x * 128 + wave * 16;
    bf16* myP = Pl + wave * (16 * 32);

    // Q fragments for 16 rows, K dim = 64 -> two 32-wide steps
    const bf16* qrow = Q + (size_t)(q0 + r16) * NHID + head * DKH;
    v16bf qa0 = ldfrag(qrow + 0 + halfl * 8, qrow + 16 + halfl * 8);
    v16bf qa1 = ldfrag(qrow + 32 + halfl * 8, qrow + 48 + halfl * 8);

    const v8f zero = {};
    v8f O[4];
#pragma unroll
    for (int j = 0; j < 4; ++j) O[j] = zero;
    float m[8], ssum[8];
#pragma unroll
    for (int r = 0; r < 8; ++r) { m[r] = -1e30f; ssum[r] = 0.0f; }

    for (int kk = 0; kk < n; kk += 32) {
#if HAVE_TDM
        if (wave == 0) {   // K tile [32 keys x 64 d] via Tensor Data Mover
            tdm_load_2d(Kx + (size_t)kk * NHID + head * DKH,
                        (unsigned)(uintptr_t)(void*)Kl, 64, 32, NHID);
            __builtin_amdgcn_s_wait_tensorcnt(0);
        }
        {   // V tile transposed manually (TDM cannot transpose)
            const int row = tid >> 3, grp = tid & 7;
            union { uint4 u; bf16 e[8]; } t;
            t.u = *reinterpret_cast<const uint4*>(
                V + (size_t)(kk + row) * NHID + head * DKH + grp * 8);
#pragma unroll
            for (int i = 0; i < 8; ++i) Vl[(grp * 8 + i) * 32 + row] = t.e[i];
        }
#else
        {   // cooperative stage of K tile (as-is) and V tile (transposed)
            const int row = tid >> 3, grp = tid & 7;
            *reinterpret_cast<uint4*>(Kl + row * 64 + grp * 8) =
                *reinterpret_cast<const uint4*>(
                    Kx + (size_t)(kk + row) * NHID + head * DKH + grp * 8);
            union { uint4 u; bf16 e[8]; } t;
            t.u = *reinterpret_cast<const uint4*>(
                V + (size_t)(kk + row) * NHID + head * DKH + grp * 8);
#pragma unroll
            for (int i = 0; i < 8; ++i) Vl[(grp * 8 + i) * 32 + row] = t.e[i];
        }
#endif
        __syncthreads();

        // S tile: 16 q-rows x 32 keys
        v8f s0 = zero, s1 = zero;
#pragma unroll
        for (int ds = 0; ds < 2; ++ds) {
            v16bf qa = ds ? qa1 : qa0;
            const bf16* b0p = Kl + r16 * 64 + ds * 32 + halfl * 8;
            const bf16* b1p = Kl + (16 + r16) * 64 + ds * 32 + halfl * 8;
            s0 = wmma_bf16(qa, ldfrag(b0p, b0p + 16), s0);
            s1 = wmma_bf16(qa, ldfrag(b1p, b1p + 16), s1);
        }

        // online softmax (row stats live across the 16 lanes of each half)
        float cm[8], corr[8], rsum[8];
#pragma unroll
        for (int r = 0; r < 8; ++r) {
            s0[r] *= 0.125f;  s1[r] *= 0.125f;      // 1/sqrt(64)
            cm[r] = fmaxf(s0[r], s1[r]);
        }
#pragma unroll
        for (int off = 8; off >= 1; off >>= 1)
#pragma unroll
            for (int r = 0; r < 8; ++r)
                cm[r] = fmaxf(cm[r], __shfl_xor(cm[r], off, 16));
#pragma unroll
        for (int r = 0; r < 8; ++r) {
            const float nm = fmaxf(m[r], cm[r]);
            corr[r] = __expf(m[r] - nm);
            m[r] = nm;
            s0[r] = __expf(s0[r] - nm);
            s1[r] = __expf(s1[r] - nm);
            rsum[r] = s0[r] + s1[r];
        }
#pragma unroll
        for (int off = 8; off >= 1; off >>= 1)
#pragma unroll
            for (int r = 0; r < 8; ++r)
                rsum[r] += __shfl_xor(rsum[r], off, 16);
#pragma unroll
        for (int r = 0; r < 8; ++r) ssum[r] = ssum[r] * corr[r] + rsum[r];
#pragma unroll
        for (int j = 0; j < 4; ++j)
#pragma unroll
            for (int r = 0; r < 8; ++r) O[j][r] *= corr[r];

        // P: C-layout -> LDS -> reload as A fragment (wave-local, LDS in-order)
#pragma unroll
        for (int r = 0; r < 8; ++r) {
            myP[(halfl * 8 + r) * 32 + r16]      = (bf16)s0[r];
            myP[(halfl * 8 + r) * 32 + 16 + r16] = (bf16)s1[r];
        }
        v16bf pa = ldfrag(myP + r16 * 32 + halfl * 8,
                          myP + r16 * 32 + 16 + halfl * 8);
#pragma unroll
        for (int j = 0; j < 4; ++j) {
            const bf16* bvp = Vl + (j * 16 + r16) * 32 + halfl * 8;
            O[j] = wmma_bf16(pa, ldfrag(bvp, bvp + 16), O[j]);
        }
        __syncthreads();
    }

    // epilogue: normalize + blend with GV, write bf16 ctx
#pragma unroll
    for (int j = 0; j < 4; ++j) {
#pragma unroll
        for (int r = 0; r < 8; ++r) {
            const int row = q0 + halfl * 8 + r;
            const int col = head * DKH + j * 16 + r16;
            const float o = O[j][r] / ssum[r];
            const float g = GV[(size_t)row * NHID + col];
            ctxb[(size_t)row * NHID + col] = (bf16)(0.5f * o + 0.5f * g);
        }
    }
}

// ---------------------------------------------------------------------------
// Residual + LayerNorm + PReLU (f32), also emits bf16 copy for next layer.
// ---------------------------------------------------------------------------
__global__ void __launch_bounds__(256)
ln_prelu_kernel(const float* __restrict__ pre, float* __restrict__ x,
                const float* __restrict__ g, const float* __restrict__ b,
                const float* __restrict__ aptr, bf16* __restrict__ xb) {
    __shared__ float rs[256], rq[256];
    const int row = blockIdx.x, t = threadIdx.x;
    const size_t base = (size_t)row * NHID;
    const float v0 = pre[base + t] + x[base + t];
    const float v1 = pre[base + 256 + t] + x[base + 256 + t];
    rs[t] = v0 + v1;
    rq[t] = v0 * v0 + v1 * v1;
    __syncthreads();
    for (int s2 = 128; s2 > 0; s2 >>= 1) {
        if (t < s2) { rs[t] += rs[t + s2]; rq[t] += rq[t + s2]; }
        __syncthreads();
    }
    const float mean = rs[0] * (1.0f / NHID);
    const float var  = rq[0] * (1.0f / NHID) - mean * mean;
    const float rstd = rsqrtf(var + 1e-5f);
    const float a = aptr[0];
    float y0 = (v0 - mean) * rstd * g[t] + b[t];
    float y1 = (v1 - mean) * rstd * g[256 + t] + b[256 + t];
    y0 = y0 >= 0.0f ? y0 : a * y0;
    y1 = y1 >= 0.0f ? y1 : a * y1;
    x[base + t] = y0;          x[base + 256 + t] = y1;
    xb[base + t] = (bf16)y0;   xb[base + 256 + t] = (bf16)y1;
}

// ---------------------------------------------------------------------------
// Classifier + log_softmax (tiny: 512x16 per row, VALU is fine)
// ---------------------------------------------------------------------------
__global__ void __launch_bounds__(128)
cls_kernel(const float* __restrict__ x, const float* __restrict__ wc,
           const float* __restrict__ bc, float* __restrict__ out) {
    __shared__ float red[128 * 16];
    const int row = blockIdx.x, t = threadIdx.x;
    float acc[16];
#pragma unroll
    for (int c = 0; c < 16; ++c) acc[c] = 0.0f;
    for (int f = t; f < NHID; f += 128) {
        const float xv = x[(size_t)row * NHID + f];
        const float* wr = wc + (size_t)f * NCLASS;
#pragma unroll
        for (int c = 0; c < 16; ++c) acc[c] += xv * wr[c];
    }
#pragma unroll
    for (int c = 0; c < 16; ++c) red[t * 16 + c] = acc[c];
    __syncthreads();
    for (int s2 = 64; s2 > 0; s2 >>= 1) {
        if (t < s2) {
#pragma unroll
            for (int c = 0; c < 16; ++c) red[t * 16 + c] += red[(t + s2) * 16 + c];
        }
        __syncthreads();
    }
    if (t == 0) {
        float lg[16], mx = -1e30f;
        for (int c = 0; c < 16; ++c) { lg[c] = red[c] + bc[c]; mx = fmaxf(mx, lg[c]); }
        float se = 0.0f;
        for (int c = 0; c < 16; ++c) se += __expf(lg[c] - mx);
        const float lse = mx + __logf(se);
        for (int c = 0; c < 16; ++c) out[(size_t)row * NCLASS + c] = lg[c] - lse;
    }
}

// ---------------------------------------------------------------------------
// Degree / conversion helpers
// ---------------------------------------------------------------------------
__global__ void edge_scale_kernel(const float* __restrict__ H, float* __restrict__ sE) {
    const int e = blockIdx.x * blockDim.x + threadIdx.x;
    if (e >= E_EDGES) return;
    float s = 0.0f;
    for (int n2 = 0; n2 < N_NODES; ++n2) s += H[(size_t)n2 * E_EDGES + e];
    sE[e] = rsqrtf(s);   // sqrt(1/DE) folded symmetrically into Ahat
}

__global__ void __launch_bounds__(256)
node_scale_kernel(const float* __restrict__ H, float* __restrict__ dv2) {
    __shared__ float red[256];
    const int row = blockIdx.x, t = threadIdx.x;
    float s = 0.0f;
    for (int e = t; e < E_EDGES; e += 256) s += H[(size_t)row * E_EDGES + e];
    red[t] = s;
    __syncthreads();
    for (int s2 = 128; s2 > 0; s2 >>= 1) {
        if (t < s2) red[t] += red[t + s2];
        __syncthreads();
    }
    if (t == 0) dv2[row] = (row == 0) ? 1.0f : rsqrtf(red[0]);  // DV2[0]=1 quirk
}

__global__ void ahat_kernel(const float* __restrict__ H, const float* __restrict__ sE,
                            bf16* __restrict__ Ahat) {
    const size_t i = (size_t)blockIdx.x * blockDim.x + threadIdx.x;
    const int e = (int)(i & (E_EDGES - 1));
    Ahat[i] = (bf16)(H[i] * sE[e]);
}

__global__ void to_bf16_kernel(const float* __restrict__ in, bf16* __restrict__ out,
                               size_t nelem) {
    const size_t i = (size_t)blockIdx.x * blockDim.x + threadIdx.x;
    if (i < nelem) out[i] = (bf16)in[i];
}

// 512x512 f32 -> transposed bf16 (weights -> [out,in] so GEMMs are A*B^T)
__global__ void transpose512_kernel(const float* __restrict__ in, bf16* __restrict__ out) {
    const int i = blockIdx.x * blockDim.x + threadIdx.x;
    const int r = i >> 9, c = i & 511;
    out[c * 512 + r] = (bf16)in[i];
}

// ---------------------------------------------------------------------------
extern "C" void kernel_launch(void* const* d_in, const int* in_sizes, int n_in,
                              void* d_out, int out_size, void* d_ws, size_t ws_size,
                              hipStream_t stream) {
    (void)in_sizes; (void)n_in; (void)out_size; (void)ws_size;
    const float* X0      = (const float*)d_in[0];
    const float* H       = (const float*)d_in[1];
    const float* w_feat  = (const float*)d_in[2];
    const float* b_feat  = (const float*)d_in[3];
    const float* Wq      = (const float*)d_in[4];
    const float* bq      = (const float*)d_in[5];
    const float* Wk      = (const float*)d_in[6];
    const float* bk      = (const float*)d_in[7];
    const float* Wv      = (const float*)d_in[8];
    const float* bv      = (const float*)d_in[9];
    const float* Wo      = (const float*)d_in[10];
    const float* bo      = (const float*)d_in[11];
    const float* ln_g    = (const float*)d_in[12];
    const float* ln_b    = (const float*)d_in[13];
    const float* prelu_a = (const float*)d_in[14];
    const float* w_cls   = (const float*)d_in[15];
    const float* b_cls   = (const float*)d_in[16];
    float* out = (float*)d_out;

    // Workspace bump allocator (~127 MB total)
    char* p = (char*)d_ws;
    auto alloc = [&](size_t bytes) -> char* {
        char* q = p;
        p += (bytes + 255) & ~(size_t)255;
        return q;
    };
    const size_t NE = (size_t)N_NODES * E_EDGES;
    const size_t NH = (size_t)N_NODES * NHID;
    const size_t WSZ = (size_t)512 * 512;

    bf16*  Ahat = (bf16*)alloc(NE * 2);
    bf16*  Gb   = (bf16*)alloc(NE * 2);
    float* sE   = (float*)alloc(E_EDGES * 4);
    float* dv2  = (float*)alloc(N_NODES * 4);
    bf16*  wfT  = (bf16*)alloc(WSZ * 2);
    bf16*  WqT  = (bf16*)alloc(NLAYER * WSZ * 2);
    bf16*  WkT  = (bf16*)alloc(NLAYER * WSZ * 2);
    bf16*  WvT  = (bf16*)alloc(NLAYER * WSZ * 2);
    bf16*  WoT  = (bf16*)alloc(NLAYER * WSZ * 2);
    bf16*  X0b  = (bf16*)alloc(NH * 2);
    float* xf   = (float*)alloc(NH * 4);
    bf16*  xb   = (bf16*)alloc(NH * 2);
    bf16*  qb   = (bf16*)alloc(NH * 2);
    bf16*  kb   = (bf16*)alloc(NH * 2);
    bf16*  vb   = (bf16*)alloc(NH * 2);
    float* GVf  = (float*)alloc(NH * 4);
    bf16*  ctxb = (bf16*)alloc(NH * 2);
    float* pre  = (float*)alloc(NH * 4);

    // ---- preprocessing ----
    edge_scale_kernel<<<E_EDGES / 256, 256, 0, stream>>>(H, sE);
    node_scale_kernel<<<N_NODES, 256, 0, stream>>>(H, dv2);
    ahat_kernel<<<(int)(NE / 256), 256, 0, stream>>>(H, sE, Ahat);
    to_bf16_kernel<<<(int)(NH / 256), 256, 0, stream>>>(X0, X0b, NH);
    transpose512_kernel<<<1024, 256, 0, stream>>>(w_feat, wfT);
    for (int l = 0; l < NLAYER; ++l) {
        transpose512_kernel<<<1024, 256, 0, stream>>>(Wq + l * WSZ, WqT + l * WSZ);
        transpose512_kernel<<<1024, 256, 0, stream>>>(Wk + l * WSZ, WkT + l * WSZ);
        transpose512_kernel<<<1024, 256, 0, stream>>>(Wv + l * WSZ, WvT + l * WSZ);
        transpose512_kernel<<<1024, 256, 0, stream>>>(Wo + l * WSZ, WoT + l * WSZ);
    }

    // ---- G = dv2 (x) dv2 * (Ahat Ahat^T), symmetric, bf16 output ----
    gemm_bf16_kernel<true><<<dim3(32, 32), 256, 0, stream>>>(
        Ahat, Ahat, nullptr, Gb, N_NODES, N_NODES, E_EDGES,
        E_EDGES, E_EDGES, N_NODES, nullptr, dv2, dv2);

    // ---- x = X0 @ w_feat + b_feat ----
    gemm_bf16_kernel<true><<<dim3(4, 32), 256, 0, stream>>>(
        X0b, wfT, xf, xb, N_NODES, NHID, NHID, NHID, NHID, NHID,
        b_feat, nullptr, nullptr);

    // ---- encoder layers ----
    for (int l = 0; l < NLAYER; ++l) {
        const size_t wo = (size_t)l * WSZ;
        gemm_bf16_kernel<true><<<dim3(4, 32), 256, 0, stream>>>(
            xb, WqT + wo, nullptr, qb, N_NODES, NHID, NHID, NHID, NHID, NHID,
            bq + l * NHID, nullptr, nullptr);
        gemm_bf16_kernel<true><<<dim3(4, 32), 256, 0, stream>>>(
            xb, WkT + wo, nullptr, kb, N_NODES, NHID, NHID, NHID, NHID, NHID,
            bk + l * NHID, nullptr, nullptr);
        gemm_bf16_kernel<true><<<dim3(4, 32), 256, 0, stream>>>(
            xb, WvT + wo, nullptr, vb, N_NODES, NHID, NHID, NHID, NHID, NHID,
            bv + l * NHID, nullptr, nullptr);

        // GV = G @ V  (B = [K,N] row-major -> TRANSB=0 path)
        gemm_bf16_kernel<false><<<dim3(4, 32), 256, 0, stream>>>(
            Gb, vb, GVf, nullptr, N_NODES, NHID, N_NODES,
            N_NODES, NHID, NHID, nullptr, nullptr, nullptr);

        // ctx = 0.5*softmax(QK^T/8)@V + 0.5*GV  (per head, flash style)
        flash_blend_kernel<<<dim3(32, NHEAD), 256, 0, stream>>>(
            qb, kb, vb, GVf, ctxb, N_NODES);

        // pre = ctx @ Wo + bo
        gemm_bf16_kernel<true><<<dim3(4, 32), 256, 0, stream>>>(
            ctxb, WoT + wo, pre, nullptr, N_NODES, NHID, NHID, NHID, NHID, NHID,
            bo + l * NHID, nullptr, nullptr);

        // x = PReLU(LN(pre + x))
        ln_prelu_kernel<<<N_NODES, 256, 0, stream>>>(
            pre, xf, ln_g + l * NHID, ln_b + l * NHID, prelu_a + l, xb);
    }

    // ---- classifier + log_softmax ----
    cls_kernel<<<N_NODES, 128, 0, stream>>>(xf, w_cls, b_cls, out);
}
